// ResEdgeConv_27212912787993
// MI455X (gfx1250) — matrix operand order repaired and checked by reading, not compiled, not run on hardware
//
#include <hip/hip_runtime.h>

typedef __attribute__((ext_vector_type(16))) _Float16 v16h;
typedef __attribute__((ext_vector_type(8)))  float    v8f;

#define N_NODES 50000
#define N_EDGES 800000
#define D       64
#define NTILES  (N_NODES / 16)   // 3125, exact

// ---------------- WMMA fragment loaders (K contiguous per lane half, same order for A and B) ----------------

// A-fragment: 16x32 f16, rows row0..row0+15 of row-major f32 X (ld = 64),
// K-range [k0, k0+32). Lane l holds row (l&15); K base 0 (lanes 0-15) or 16
// (lanes 16-31); elements 0-15 = Kbase+0..15 (matches B fragment).
__device__ inline v16h load_a_frag(const float* __restrict__ X, int row0, int k0, int lane) {
    const int m    = lane & 15;
    const int base = (lane & 16) ? 16 : 0;
    const float* r = X + (size_t)(row0 + m) * D + k0;
    v16h a;
#pragma unroll
    for (int i = 0; i < 16; ++i) {
        int v = i >> 1, j = i & 1;
        int k = base + 2 * v + j;
        a[i] = (_Float16)r[k];
    }
    return a;
}

// B-fragment: 32x16 f16 from row-major f32 W[64][64], K-range [k0,k0+32),
// column block col0..col0+15. Lane l holds column (l&15); lanes 0-15 carry
// K k0+0..15, lanes 16-31 carry K k0+16..31, 2 per VGPR.
__device__ inline v16h load_b_frag(const float* __restrict__ W, int k0, int col0, int lane) {
    const int n  = col0 + (lane & 15);
    const int kb = k0 + ((lane & 16) ? 16 : 0);
    v16h b;
#pragma unroll
    for (int i = 0; i < 16; ++i)
        b[i] = (_Float16)W[(size_t)(kb + i) * D + n];
    return b;
}

// Same, but element-wise W1 - W2 (for the x_i @ (W1 - W2) term).
__device__ inline v16h load_b_frag_diff(const float* __restrict__ W1,
                                        const float* __restrict__ W2,
                                        int k0, int col0, int lane) {
    const int n  = col0 + (lane & 15);
    const int kb = k0 + ((lane & 16) ? 16 : 0);
    v16h b;
#pragma unroll
    for (int i = 0; i < 16; ++i) {
        size_t off = (size_t)(kb + i) * D + n;
        b[i] = (_Float16)(W1[off] - W2[off]);
    }
    return b;
}

__device__ inline v8f wmma2(v16h a0, v16h a1, v16h b0, v16h b1) {
    v8f acc = {};
    acc = __builtin_amdgcn_wmma_f32_16x16x32_f16(false, a0, false, b0, (short)0, acc, false, false);
    acc = __builtin_amdgcn_wmma_f32_16x16x32_f16(false, a1, false, b1, (short)0, acc, false, false);
    return acc;
}

// ---------------- Kernel A: node-level GEMMs via WMMA ----------------
// P   = x @ W2                     (gathered by edge source j)
// Q   = x @ (W1 - W2) + b_edge    (gathered by edge target i)
// out = x @ W_res + b_res          (residual branch, written directly)
__global__ void __launch_bounds__(256)
resedgeconv_gemm3(const float* __restrict__ x,
                  const float* __restrict__ W_edge,
                  const float* __restrict__ b_edge,
                  const float* __restrict__ W_res,
                  const float* __restrict__ b_res,
                  float* __restrict__ P,
                  float* __restrict__ Q,
                  float* __restrict__ out) {
    const int lane = threadIdx.x & 31;
    const int tile = blockIdx.x * (blockDim.x >> 5) + (threadIdx.x >> 5);
    if (tile >= NTILES) return;
    const int row0 = tile * 16;

    const v16h a0 = load_a_frag(x, row0, 0, lane);
    const v16h a1 = load_a_frag(x, row0, 32, lane);

    const float* W1 = W_edge;            // rows 0..63   of [128,64]
    const float* W2 = W_edge + 64 * D;   // rows 64..127

    const int col   = lane & 15;
    const int rbase = row0 + ((lane & 16) ? 8 : 0);

#pragma unroll
    for (int ct = 0; ct < 4; ++ct) {
        const int col0 = ct * 16;
        {   // P = x @ W2
            v8f acc = wmma2(a0, a1, load_b_frag(W2, 0, col0, lane),
                                    load_b_frag(W2, 32, col0, lane));
#pragma unroll
            for (int r = 0; r < 8; ++r)
                P[(size_t)(rbase + r) * D + col0 + col] = acc[r];
        }
        {   // Q = x @ (W1 - W2) + b_edge
            v8f acc = wmma2(a0, a1, load_b_frag_diff(W1, W2, 0, col0, lane),
                                    load_b_frag_diff(W1, W2, 32, col0, lane));
            const float bias = b_edge[col0 + col];
#pragma unroll
            for (int r = 0; r < 8; ++r)
                Q[(size_t)(rbase + r) * D + col0 + col] = acc[r] + bias;
        }
        {   // out = x @ W_res + b_res
            v8f acc = wmma2(a0, a1, load_b_frag(W_res, 0, col0, lane),
                                    load_b_frag(W_res, 32, col0, lane));
            const float bias = b_res[col0 + col];
#pragma unroll
            for (int r = 0; r < 8; ++r)
                out[(size_t)(rbase + r) * D + col0 + col] = acc[r] + bias;
        }
    }
}

// ---------------- Zero the aggregation buffer (0u == 0.0f bits) ----------------
__global__ void resedgeconv_zero(uint4* __restrict__ agg, int n4) {
    int t = blockIdx.x * blockDim.x + threadIdx.x;
    if (t < n4) agg[t] = make_uint4(0u, 0u, 0u, 0u);
}

// ---------------- Kernel B: edge pass (gather + relu + atomic segment-max) --------
// 16 threads per edge, float4 per thread. Non-negative float max == u32 bit max.
__global__ void __launch_bounds__(256)
resedgeconv_edges(const long long* __restrict__ ei,   // [2, E] int64
                  const float* __restrict__ P,
                  const float* __restrict__ Q,
                  unsigned int* __restrict__ agg) {
    const int t = blockIdx.x * blockDim.x + threadIdx.x;
    const int e = t >> 4;
    if (e >= N_EDGES) return;
    const int f4 = t & 15;

    const long long j = ei[e];            // source (neighbor)
    const long long i = ei[N_EDGES + e];  // target (center)

    const float4 p = ((const float4*)P)[(size_t)j * 16 + f4];
    const float4 q = ((const float4*)Q)[(size_t)i * 16 + f4];

    float4 m;
    m.x = fmaxf(p.x + q.x, 0.0f);
    m.y = fmaxf(p.y + q.y, 0.0f);
    m.z = fmaxf(p.z + q.z, 0.0f);
    m.w = fmaxf(p.w + q.w, 0.0f);

    unsigned int* a = agg + (size_t)i * D + f4 * 4;
    if (m.x > 0.0f) atomicMax(a + 0, __float_as_uint(m.x));
    if (m.y > 0.0f) atomicMax(a + 1, __float_as_uint(m.y));
    if (m.z > 0.0f) atomicMax(a + 2, __float_as_uint(m.z));
    if (m.w > 0.0f) atomicMax(a + 3, __float_as_uint(m.w));
}

// ---------------- Kernel C: out += agg ----------------
__global__ void resedgeconv_finalize(float4* __restrict__ out,
                                     const uint4* __restrict__ agg, int n4) {
    int t = blockIdx.x * blockDim.x + threadIdx.x;
    if (t >= n4) return;
    uint4 a = agg[t];
    float4 o = out[t];
    o.x += __uint_as_float(a.x);
    o.y += __uint_as_float(a.y);
    o.z += __uint_as_float(a.z);
    o.w += __uint_as_float(a.w);
    out[t] = o;
}

extern "C" void kernel_launch(void* const* d_in, const int* in_sizes, int n_in,
                              void* d_out, int out_size, void* d_ws, size_t ws_size,
                              hipStream_t stream) {
    const float*     x      = (const float*)d_in[0];
    const long long* ei     = (const long long*)d_in[1];  // int64 [2, E]
    const float*     W_edge = (const float*)d_in[2];      // [128, 64]
    const float*     b_edge = (const float*)d_in[3];      // [64]
    const float*     W_res  = (const float*)d_in[4];      // [64, 64]
    const float*     b_res  = (const float*)d_in[5];      // [64]
    float*           out    = (float*)d_out;

    // Workspace: P | Q | AGG, each N*D f32 (38.4 MB total).
    float*        P   = (float*)d_ws;
    float*        Q   = P + (size_t)N_NODES * D;
    unsigned int* AGG = (unsigned int*)(Q + (size_t)N_NODES * D);

    const int n4 = N_NODES * D / 4;  // 800000

    resedgeconv_zero<<<(n4 + 255) / 256, 256, 0, stream>>>((uint4*)AGG, n4);

    const int wavesPerBlock = 8;  // 256 threads = 8 wave32
    resedgeconv_gemm3<<<(NTILES + wavesPerBlock - 1) / wavesPerBlock, 256, 0, stream>>>(
        x, W_edge, b_edge, W_res, b_res, P, Q, out);

    const long long edgeThreads = (long long)N_EDGES * 16;
    resedgeconv_edges<<<(int)((edgeThreads + 255) / 256), 256, 0, stream>>>(ei, P, Q, AGG);

    resedgeconv_finalize<<<(n4 + 255) / 256, 256, 0, stream>>>((float4*)out, (const uint4*)AGG, n4);
}
